// GCN_10007273799959
// MI455X (gfx1250) — matrix operand order repaired and checked by reading, not compiled, not run on hardware
//
#include <hip/hip_runtime.h>
#include <hip/hip_bf16.h>

typedef __attribute__((ext_vector_type(16))) _Float16 v16h;
typedef __attribute__((ext_vector_type(2)))  _Float16 v2h;
typedef __attribute__((ext_vector_type(8)))  float    v8f;

#define F_IN  256
#define F_OUT 128

// ---------------------------------------------------------------------------
// GEMM: h[N,128] = x[N,256] @ W[256,128], via v_wmma_f32_16x16x32_f16.
// Block = 256 threads = 8 wave32s; each wave computes a 16-row strip (16x128).
// W is converted to f16 and staged TRANSPOSED in LDS as sWt[n][k] so each
// B-fragment half-pair (k, k+1) is one 32-bit LDS load (compiler merges the
// 8 pairs into 2x ds_load_b128). B fragments are double-buffered so the
// ds_load latency overlaps the previous WMMA instead of stalling on
// s_wait_dscnt 0 each K-step.
// ---------------------------------------------------------------------------
__global__ __launch_bounds__(256) void gcn_gemm_wmma(
    const float* __restrict__ x, const float* __restrict__ W,
    float* __restrict__ h, int nnodes)
{
    __shared__ _Float16 sWt[F_OUT * F_IN];   // 64 KB, [n][k]

    for (int idx = threadIdx.x; idx < F_IN * F_OUT; idx += 256) {
        int k = idx >> 7;       // row of W   (0..255)
        int n = idx & 127;      // col of W   (0..127)
        sWt[n * F_IN + k] = (_Float16)W[idx];
    }
    __syncthreads();

    const int lane    = threadIdx.x & 31;
    const int wave    = threadIdx.x >> 5;
    const int rowbase = blockIdx.x * 128 + wave * 16;
    const int m       = lane & 15;
    const int hi      = lane >> 4;           // 0: lanes 0-15, 1: lanes 16-31
    const int myrow   = rowbase + m;
    const bool full   = (rowbase + 16) <= nnodes;   // wave-uniform fast path

    // ---- A fragments: 8 ktiles of 16x32 f16 (ISA 16-bit A layout) ----
    // VGPR v, lane group hi: k = (v>=4?16:0) + hi*8 + (v&3)*2
    v16h a[8];
    if (full) {
        const float* xr = x + (size_t)myrow * F_IN;
        #pragma unroll
        for (int kt = 0; kt < 8; ++kt) {
            #pragma unroll
            for (int v = 0; v < 8; ++v) {
                int k = kt * 32 + ((v >> 2) * 16) + (hi * 8) + ((v & 3) * 2);
                float2 f = *reinterpret_cast<const float2*>(xr + k);
                a[kt][2 * v]     = (_Float16)f.x;
                a[kt][2 * v + 1] = (_Float16)f.y;
            }
        }
    } else {
        #pragma unroll
        for (int kt = 0; kt < 8; ++kt) {
            #pragma unroll
            for (int v = 0; v < 8; ++v) {
                int k = kt * 32 + ((v >> 2) * 16) + (hi * 8) + ((v & 3) * 2);
                float2 f = make_float2(0.0f, 0.0f);
                if (myrow < nnodes)
                    f = *reinterpret_cast<const float2*>(x + (size_t)myrow * F_IN + k);
                a[kt][2 * v]     = (_Float16)f.x;
                a[kt][2 * v + 1] = (_Float16)f.y;
            }
        }
    }

    // ---- per N-tile: 8 chained WMMAs over K (double-buffered B), store ----
    #pragma unroll 1
    for (int nt = 0; nt < 8; ++nt) {
        const int n = nt * 16 + m;
        const _Float16* wb = &sWt[n * F_IN + hi * 8];

        auto loadB = [&](int kt) {
            v16h b;
            #pragma unroll
            for (int v = 0; v < 8; ++v) {
                v2h p = *reinterpret_cast<const v2h*>(wb + kt * 32 + (v >> 2) * 16 + (v & 3) * 2);
                b[2 * v]     = p[0];
                b[2 * v + 1] = p[1];
            }
            return b;
        };

        v8f  c    = {};
        v16h bcur = loadB(0);
        #pragma unroll
        for (int kt = 0; kt < 8; ++kt) {
            v16h bnext = bcur;
            if (kt < 7) bnext = loadB(kt + 1);   // prefetch next K-step's B
            c = __builtin_amdgcn_wmma_f32_16x16x32_f16(
                    /*neg_a=*/false, a[kt], /*neg_b=*/false, bcur,
                    /*c_mod=*/(short)0, c, /*reuse_a=*/false, /*reuse_b=*/false);
            bcur = bnext;
        }

        // C/D layout: VGPR v -> row m = v + hi*8, lane -> col n
        if (full) {
            float* hp = h + (size_t)(rowbase + hi * 8) * F_OUT + n;
            #pragma unroll
            for (int v = 0; v < 8; ++v)
                hp[(size_t)v * F_OUT] = c[v];
        } else {
            #pragma unroll
            for (int v = 0; v < 8; ++v) {
                int row = rowbase + v + hi * 8;
                if (row < nnodes) h[(size_t)row * F_OUT + n] = c[v];
            }
        }
    }
}

// ---------------------------------------------------------------------------
// Degree passes
// ---------------------------------------------------------------------------
__global__ void deg_init(float* __restrict__ deg, int n) {
    int i = blockIdx.x * blockDim.x + threadIdx.x;
    if (i < n) deg[i] = 1.0f;                // self-loop contributes 1 per node
}

__global__ void deg_count(const int* __restrict__ col, float* __restrict__ deg, int E) {
    int e = blockIdx.x * blockDim.x + threadIdx.x;
    if (e < E) atomicAdd(&deg[col[e]], 1.0f);
}

__global__ void deg_rsqrt(float* __restrict__ deg, float* __restrict__ dinv, int n) {
    int i = blockIdx.x * blockDim.x + threadIdx.x;
    if (i < n) {
        float d = deg[i];
        dinv[i] = (d > 0.0f) ? rsqrtf(d) : 0.0f;
    }
}

// ---------------------------------------------------------------------------
// out init: self-loop term  out[i, f] = h[i, f] * dinv[i]^2
// (fully overwrites d_out each call -> graph-replay deterministic)
// ---------------------------------------------------------------------------
__global__ void out_init(const float* __restrict__ h, const float* __restrict__ dinv,
                         float* __restrict__ out, int total) {
    int idx = blockIdx.x * blockDim.x + threadIdx.x;
    if (idx < total) {
        int node = idx >> 7;
        float di = dinv[node];
        out[idx] = h[idx] * di * di;
    }
}

// ---------------------------------------------------------------------------
// scatter: one wave per edge; lane gathers float4 of h[row], scales by
// dinv[row]*dinv[col], atomically adds into out[col]. h and out are both
// L2-resident (51 MB each vs 192 MB L2), so this is L2-atomic bound.
// ---------------------------------------------------------------------------
__global__ void edge_scatter(const float* __restrict__ h, const float* __restrict__ dinv,
                             const int* __restrict__ rowi, const int* __restrict__ coli,
                             float* __restrict__ out, int E) {
    long long gid = (long long)blockIdx.x * blockDim.x + threadIdx.x;
    int e    = (int)(gid >> 5);
    int lane = (int)(gid & 31);
    if (e >= E) return;
    int r = rowi[e];
    int c = coli[e];
    float nrm = dinv[r] * dinv[c];
    float4 v = *reinterpret_cast<const float4*>(h + (size_t)r * F_OUT + lane * 4);
    float* op = out + (size_t)c * F_OUT + lane * 4;
    atomicAdd(op + 0, v.x * nrm);
    atomicAdd(op + 1, v.y * nrm);
    atomicAdd(op + 2, v.z * nrm);
    atomicAdd(op + 3, v.w * nrm);
}

// ---------------------------------------------------------------------------
// finalize: + bias, per-channel PReLU (in place)
// ---------------------------------------------------------------------------
__global__ void finalize(float* __restrict__ out, const float* __restrict__ bias,
                         const float* __restrict__ pa, int total) {
    int idx = blockIdx.x * blockDim.x + threadIdx.x;
    if (idx < total) {
        int f = idx & 127;
        float v = out[idx] + bias[f];
        out[idx] = (v > 0.0f) ? v : pa[f] * v;
    }
}

// ---------------------------------------------------------------------------
extern "C" void kernel_launch(void* const* d_in, const int* in_sizes, int n_in,
                              void* d_out, int out_size, void* d_ws, size_t ws_size,
                              hipStream_t stream) {
    const float* x    = (const float*)d_in[0];
    const int*   ei   = (const int*)d_in[1];        // [2, E] row-major
    const float* W    = (const float*)d_in[2];
    const float* bias = (const float*)d_in[3];
    const float* pa   = (const float*)d_in[4];

    const int nnodes = in_sizes[0] / F_IN;          // 100000
    const int E      = in_sizes[1] / 2;             // 1600000
    const int* rowi  = ei;                          // sources
    const int* coli  = ei + E;                      // targets

    // workspace layout
    char* ws = (char*)d_ws;
    float* h    = (float*)ws;                                   // N*128 f32
    float* deg  = (float*)(ws + (size_t)nnodes * F_OUT * 4);    // N f32
    float* dinv = deg + nnodes;                                 // N f32
    float* out  = (float*)d_out;

    const int total = nnodes * F_OUT;

    // 1) GEMM via WMMA
    int gemm_blocks = (nnodes + 127) / 128;
    gcn_gemm_wmma<<<gemm_blocks, 256, 0, stream>>>(x, W, h, nnodes);

    // 2) degrees
    deg_init <<<(nnodes + 255) / 256, 256, 0, stream>>>(deg, nnodes);
    deg_count<<<(E + 255) / 256,      256, 0, stream>>>(coli, deg, E);
    deg_rsqrt<<<(nnodes + 255) / 256, 256, 0, stream>>>(deg, dinv, nnodes);

    // 3) self-loop init of out
    out_init<<<(total + 255) / 256, 256, 0, stream>>>(h, dinv, out, total);

    // 4) edge scatter (one wave per edge)
    long long sc_threads = (long long)E * 32;
    int sc_blocks = (int)((sc_threads + 255) / 256);
    edge_scatter<<<sc_blocks, 256, 0, stream>>>(h, dinv, rowi, coli, out, E);

    // 5) bias + PReLU
    finalize<<<(total + 255) / 256, 256, 0, stream>>>(out, bias, pa, total);
}
